// LateInteraction_41772851921084
// MI455X (gfx1250) — matrix operand order repaired and checked by reading, not compiled, not run on hardware
//
#include <hip/hip_runtime.h>
#include <math.h>

typedef __attribute__((ext_vector_type(16))) _Float16 v16h;
typedef __attribute__((ext_vector_type(8)))  _Float16 v8h;
typedef __attribute__((ext_vector_type(4)))  _Float16 v4h;
typedef __attribute__((ext_vector_type(8)))  float    v8f;
typedef __attribute__((ext_vector_type(4)))  unsigned int u32x4;
typedef __attribute__((ext_vector_type(8)))  int          i32x8;
typedef __attribute__((ext_vector_type(4)))  int          i32x4;

#define SEQ 512
#define HID 768
#define NB  16
#define SROWS 128   // s-rows per block (8 waves x 16)
#define NTILE 32    // t-columns per outer iteration (2 x 16)

#if __has_builtin(__builtin_amdgcn_tensor_load_to_lds)
#define HAVE_TDM 1
#else
#define HAVE_TDM 0
#endif

// ---------------------------------------------------------------------------
// Kernel 1: L2-normalize each length-768 row and convert to f16.
// ---------------------------------------------------------------------------
__global__ __launch_bounds__(256) void normalize_rows_f16(
    const float* __restrict__ src, _Float16* __restrict__ dst, int nrows)
{
  const int wave = threadIdx.x >> 5;
  const int lane = threadIdx.x & 31;
  const int row  = blockIdx.x * 8 + wave;
  if (row >= nrows) return;

  const float4* s4 = (const float4*)(src + (size_t)row * HID);
  float4 v[6];
  float ss = 0.f;
#pragma unroll
  for (int k = 0; k < 6; ++k) {
    v[k] = s4[lane + 32 * k];
    ss += v[k].x * v[k].x + v[k].y * v[k].y + v[k].z * v[k].z + v[k].w * v[k].w;
  }
#pragma unroll
  for (int off = 16; off; off >>= 1) ss += __shfl_xor(ss, off, 32);

  const float inv = 1.0f / fmaxf(sqrtf(ss), 1e-12f);

  _Float16* drow = dst + (size_t)row * HID;
#pragma unroll
  for (int k = 0; k < 6; ++k) {
    v4h h;
    h[0] = (_Float16)(v[k].x * inv);
    h[1] = (_Float16)(v[k].y * inv);
    h[2] = (_Float16)(v[k].z * inv);
    h[3] = (_Float16)(v[k].w * inv);
    *(v4h*)(drow + 4 * (lane + 32 * k)) = h;
  }
}

// ---------------------------------------------------------------------------
// TDM: issue a 2-D tile load (rows x cols of f16) global -> LDS via the
// Tensor Data Mover.  Descriptor layout per CDNA5 ISA ch.8 (D# groups).
// Caller must be a single wave; EXEC is ignored by TENSOR_LOAD_TO_LDS.
// ---------------------------------------------------------------------------
#if HAVE_TDM
__device__ __forceinline__ void tdm_load_2d(const _Float16* gsrc,
                                            unsigned lds_byte_off,
                                            unsigned tile_d0,   // elems per row
                                            unsigned tile_d1,   // rows
                                            unsigned long long stride0) // elems
{
  unsigned long long ga = (unsigned long long)(uintptr_t)gsrc;
  u32x4 g0;
  g0[0] = 1u;                                   // count=1, user descriptor
  g0[1] = lds_byte_off;                         // lds_addr
  g0[2] = (unsigned)(ga & 0xFFFFFFFFu);         // global_addr[31:0]
  g0[3] = (unsigned)((ga >> 32) & 0x01FFFFFFu)  // global_addr[56:32]
        | 0x80000000u;                          // type=2 ("image")
  i32x8 g1;
  g1[0] = 0x00010000;                           // workgroup_mask=0, data_size=2B
  g1[1] = (int)((tile_d0 & 0xFFFFu) << 16);     // tensor_dim0[15:0]
  g1[2] = (int)(((tile_d0 >> 16) & 0xFFFFu) |   // tensor_dim0[31:16]
                ((tile_d1 & 0xFFFFu) << 16));   // tensor_dim1[15:0]
  g1[3] = (int)(((tile_d1 >> 16) & 0xFFFFu) |   // tensor_dim1[31:16]
                ((tile_d0 & 0xFFFFu) << 16));   // tile_dim0
  g1[4] = (int)(tile_d1 & 0xFFFFu);             // tile_dim1 (tile_dim2=0)
  g1[5] = (int)(stride0 & 0xFFFFFFFFull);       // tensor_dim0_stride[31:0]
  g1[6] = (int)((stride0 >> 32) & 0xFFFFull);   // stride0[47:32]; dim1_stride=0
  g1[7] = 0;
  i32x4 g2 = {0, 0, 0, 0};
  i32x4 g3 = {0, 0, 0, 0};
#if defined(__clang_major__) && (__clang_major__ >= 23)
  i32x8 g4 = {0, 0, 0, 0, 0, 0, 0, 0};
  __builtin_amdgcn_tensor_load_to_lds(g0, g1, g2, g3, g4, 0);
#else
  __builtin_amdgcn_tensor_load_to_lds(g0, g1, g2, g3, 0);
#endif
}
#endif

// ---------------------------------------------------------------------------
// Kernel 2: per (i,j) pair, flash-softmax over t with WMMA f16->f32 GEMM.
// Grid (16,16,4); 256 threads = 8 waves; wave owns 16 s-rows x 32 t-cols/iter.
// LDS: Q 128x768 f16 (192KB) + 2 x K 32x768 f16 (96KB) + k_mask (2KB) = 290KB.
// K tiles streamed by the Tensor Data Mover, double-buffered on TENSORcnt.
// ---------------------------------------------------------------------------
__global__ __launch_bounds__(256) void late_interaction_wmma(
    const _Float16* __restrict__ Qh, const _Float16* __restrict__ Kh,
    const float* __restrict__ q_mask, const float* __restrict__ k_mask,
    const float* __restrict__ logit_scale_p, const float* __restrict__ alpha_p,
    float* __restrict__ out)
{
  extern __shared__ _Float16 smem[];
  _Float16* q_lds  = smem;                               // SROWS*HID
  _Float16* k_lds  = smem + SROWS * HID;                 // 2 * NTILE*HID
  float*    km_lds = (float*)(smem + (SROWS + 2 * NTILE) * HID);

  const int i       = blockIdx.x;
  const int j       = blockIdx.y;
  const int s_block = blockIdx.z * SROWS;

  const int tid  = threadIdx.x;
  const int lane = tid & 31;
  const int wave = tid >> 5;
  const int hi   = lane >> 4;
  const int nl   = lane & 15;

  const float scale = __expf(logit_scale_p[0]);
  const float ar    = alpha_p[0];
  const float alpha = ar > 0.f ? ar : 0.01f * ar;        // leaky_relu

  const _Float16* qsrc  = Qh + ((size_t)i * SEQ + s_block) * HID;
  const _Float16* kbase = Kh + (size_t)j * SEQ * HID;

#if HAVE_TDM
  const unsigned dynbase = __builtin_amdgcn_groupstaticsize();
  const unsigned q_off   = dynbase;
  const unsigned k_off0  = dynbase + SROWS * HID * 2;
  const unsigned k_off1  = k_off0 + NTILE * HID * 2;
  if (wave == 0) {
    tdm_load_2d(qsrc,  q_off,  HID, SROWS, HID);         // whole Q chunk
    tdm_load_2d(kbase, k_off0, HID, NTILE, HID);         // first K tile
  }
#else
  {
    const v8h* qs = (const v8h*)qsrc;
    v8h* qd = (v8h*)q_lds;
    for (int idx = tid; idx < SROWS * HID / 8; idx += 256) qd[idx] = qs[idx];
    const v8h* ks = (const v8h*)kbase;
    v8h* kd = (v8h*)k_lds;
    for (int idx = tid; idx < NTILE * HID / 8; idx += 256) kd[idx] = ks[idx];
  }
#endif
  for (int t = tid; t < SEQ; t += 256) km_lds[t] = k_mask[j * SEQ + t];

  const int row_local = wave * 16 + nl;   // A-matrix row for this lane

  // Online softmax state: reg r holds row (wave*16 + r + hi*8)
  float m[8], Z[8], Acc[8];
#pragma unroll
  for (int r = 0; r < 8; ++r) { m[r] = -1e30f; Z[r] = 0.f; Acc[r] = 0.f; }

  for (int t00 = 0; t00 < SEQ; t00 += NTILE) {
    const int buf = (t00 / NTILE) & 1;
#if HAVE_TDM
    if (wave == 0) __builtin_amdgcn_s_wait_tensorcnt(0); // current buf ready
#endif
    __syncthreads();                                     // publish; all readers done
#if HAVE_TDM
    if (wave == 0 && t00 + NTILE < SEQ)
      tdm_load_2d(kbase + (size_t)(t00 + NTILE) * HID,
                  buf ? k_off0 : k_off1, HID, NTILE, HID);
#else
    if (t00 + NTILE < SEQ) {
      const v8h* ks = (const v8h*)(kbase + (size_t)(t00 + NTILE) * HID);
      v8h* kd = (v8h*)(k_lds + (buf ? 0 : NTILE * HID));
      for (int idx = tid; idx < NTILE * HID / 8; idx += 256) kd[idx] = ks[idx];
    }
#endif
    const _Float16* kbuf = k_lds + (buf ? NTILE * HID : 0);

    v8f zero = {};
    v8f cc[2]; cc[0] = zero; cc[1] = zero;               // 16x32 accumulators

    for (int kk = 0; kk < HID; kk += 32) {
      // A fragment (16x32 f16), reused for both column tiles
      const _Float16* qa = q_lds + row_local * HID + kk + hi * 8;
      v8h a0 = *(const v8h*)qa;
      v8h a1 = *(const v8h*)(qa + 16);
      v16h a;
#pragma unroll
      for (int e = 0; e < 8; ++e) { a[e] = a0[e]; a[e + 8] = a1[e]; }

      // Two B fragments (32x16 f16): columns nl and 16+nl
      const _Float16* kb0 = kbuf + nl * HID        + kk + hi * 8;
      const _Float16* kb1 = kbuf + (16 + nl) * HID + kk + hi * 8;
      v8h b00 = *(const v8h*)kb0, b01 = *(const v8h*)(kb0 + 16);
      v8h b10 = *(const v8h*)kb1, b11 = *(const v8h*)(kb1 + 16);
      v16h b0, b1;
#pragma unroll
      for (int e = 0; e < 8; ++e) {
        b0[e] = b00[e]; b0[e + 8] = b01[e];
        b1[e] = b10[e]; b1[e + 8] = b11[e];
      }
      cc[0] = __builtin_amdgcn_wmma_f32_16x16x32_f16(
                  false, a, false, b0, (short)0, cc[0], false, false);
      cc[1] = __builtin_amdgcn_wmma_f32_16x16x32_f16(
                  false, a, false, b1, (short)0, cc[1], false, false);
    }

    // ---- online softmax update for both 16-column tiles ----
#pragma unroll
    for (int nt = 0; nt < 2; ++nt) {
      const int   t   = t00 + nt * 16 + nl;
      const float kmv = km_lds[t];
      if (kmv > 0.f) {
#pragma unroll
        for (int r = 0; r < 8; ++r) {
          const int   s   = s_block + wave * 16 + r + hi * 8;
          const float S   = cc[nt][r];
          const float wgt = __expf(-alpha * fabsf((float)(s - t)));
          const float l   = scale * wgt * S;
          const float nm  = fmaxf(m[r], l);
          const float co  = __expf(m[r] - nm);  // 0 on first valid (m=-1e30)
          const float e   = __expf(l - nm);
          Z[r]   = Z[r]   * co + e;
          Acc[r] = Acc[r] * co + e * S;
          m[r]   = nm;
        }
      }
    }
  }

  // ---- merge the 16 column-lanes of each half-wave ----
#pragma unroll
  for (int off = 1; off < 16; off <<= 1) {
#pragma unroll
    for (int r = 0; r < 8; ++r) {
      const float m2 = __shfl_xor(m[r],   off, 16);
      const float Z2 = __shfl_xor(Z[r],   off, 16);
      const float A2 = __shfl_xor(Acc[r], off, 16);
      const float nm = fmaxf(m[r], m2);
      const float c1 = __expf(m[r] - nm);
      const float c2 = __expf(m2   - nm);
      Z[r]   = Z[r]   * c1 + Z2 * c2;
      Acc[r] = Acc[r] * c1 + A2 * c2;
      m[r]   = nm;
    }
  }

  if (nl == 0) {
    float part = 0.f;
#pragma unroll
    for (int r = 0; r < 8; ++r) {
      const int   s   = s_block + wave * 16 + r + hi * 8;
      const float sc  = (Z[r] > 0.f) ? (Acc[r] / Z[r]) : 0.f;  // nan_to_num
      const float qmv = q_mask[i * SEQ + s];
      part += (qmv > 0.f) ? sc * qmv : 0.f;
    }
    atomicAdd(&out[i * NB + j], part);
  }
}

// ---------------------------------------------------------------------------
extern "C" void kernel_launch(void* const* d_in, const int* in_sizes, int n_in,
                              void* d_out, int out_size, void* d_ws, size_t ws_size,
                              hipStream_t stream) {
  const float* q_embs = (const float*)d_in[0];
  const float* k_embs = (const float*)d_in[1];
  const float* q_mask = (const float*)d_in[2];
  const float* k_mask = (const float*)d_in[3];
  const float* lscale = (const float*)d_in[4];
  const float* araw   = (const float*)d_in[5];
  float* out = (float*)d_out;

  _Float16* Qh = (_Float16*)d_ws;
  _Float16* Kh = Qh + (size_t)NB * SEQ * HID;

  hipMemsetAsync(d_out, 0, (size_t)NB * NB * sizeof(float), stream);

  normalize_rows_f16<<<NB * SEQ / 8, 256, 0, stream>>>(q_embs, Qh, NB * SEQ);
  normalize_rows_f16<<<NB * SEQ / 8, 256, 0, stream>>>(k_embs, Kh, NB * SEQ);

  const size_t lds_bytes = (size_t)(SROWS + 2 * NTILE) * HID * sizeof(_Float16)
                         + (size_t)SEQ * sizeof(float);   // 296,960 B (< 320KB WGP LDS)
  hipFuncSetAttribute((const void*)late_interaction_wmma,
                      hipFuncAttributeMaxDynamicSharedMemorySize, (int)lds_bytes);

  late_interaction_wmma<<<dim3(NB, NB, 4), 256, lds_bytes, stream>>>(
      Qh, Kh, q_mask, k_mask, lscale, araw, out);
}
